// AttLayer_43035572306647
// MI455X (gfx1250) — matrix operand order repaired and checked by reading, not compiled, not run on hardware
//
#include <hip/hip_runtime.h>

typedef __attribute__((ext_vector_type(16))) _Float16 v16h;
typedef __attribute__((ext_vector_type(8)))  _Float16 v8h;
typedef __attribute__((ext_vector_type(8)))  float    v8f;

#define NHEAD    16
#define HEAD_DIM 64
#define BATCH    2
#define SEQ      2048
#define DM       1024          // d_model == att_model
#define MROWS    (BATCH*SEQ)   // 4096

#if __has_builtin(__builtin_amdgcn_global_load_async_to_lds_b128)
#define HAS_ASYNC 1
#else
#define HAS_ASYNC 0
#endif

typedef int v4i __attribute__((vector_size(4 * sizeof(int))));
typedef __attribute__((address_space(1))) v4i* gv4i_p;   // global ("__device__")
typedef __attribute__((address_space(3))) v4i* lv4i_p;   // LDS

static __device__ __forceinline__ void wait_async0() {
#if __has_builtin(__builtin_amdgcn_s_wait_asynccnt)
  __builtin_amdgcn_s_wait_asynccnt(0);
#else
  asm volatile("s_wait_asynccnt 0x0" ::: "memory");
#endif
}

#if HAS_ASYNC
// per-lane 16B global -> LDS, tracked by ASYNCcnt
static __device__ __forceinline__ void async_copy16(const _Float16* g, _Float16* l) {
  __builtin_amdgcn_global_load_async_to_lds_b128(
      (gv4i_p)(unsigned long long)g,
      (lv4i_p)(unsigned int)(unsigned long long)l, 0, 0);
}
#endif

// raw v_exp_f32 (underflow flushes to zero -- exactly what softmax wants)
static __device__ __forceinline__ float fast_exp2(float x) {
#if __has_builtin(__builtin_amdgcn_exp2f)
  return __builtin_amdgcn_exp2f(x);
#else
  return exp2f(x);
#endif
}

static __device__ __forceinline__ v16h ld16(const _Float16* p0, const _Float16* p1) {
  v8h a = *(const v8h*)p0;
  v8h b = *(const v8h*)p1;
  return __builtin_shufflevector(a, b, 0,1,2,3,4,5,6,7,8,9,10,11,12,13,14,15);
}

// ---------------------------------------------------------------- stage 0
__global__ __launch_bounds__(256) void cvt_f32_to_f16(const float* __restrict__ in,
                                                      _Float16* __restrict__ out, int n) {
  for (int i = blockIdx.x * blockDim.x + threadIdx.x; i < n; i += gridDim.x * blockDim.x)
    out[i] = (_Float16)in[i];
}

// W[K,N] fp32 -> Wt[N,K] f16 (32x32 LDS tiles)
__global__ __launch_bounds__(256) void transpose_w(const float* __restrict__ W,
                                                   _Float16* __restrict__ Wt) {
  __shared__ float tile[32][33];
  const int kt = blockIdx.y * 32, nt = blockIdx.x * 32;
  const int tx = threadIdx.x, ty = threadIdx.y;   // (32,8)
  for (int i = 0; i < 32; i += 8)
    tile[ty + i][tx] = W[(size_t)(kt + ty + i) * DM + nt + tx];
  __syncthreads();
  for (int i = 0; i < 32; i += 8)
    Wt[(size_t)(nt + ty + i) * DM + kt + tx] = (_Float16)tile[tx][ty + i];
}

// ---------------------------------------------------------------- GEMM (NT)
// C[4096,1024] = X[4096,1024] @ Wt[1024,1024]^T + bias
// MODE 0: fp32 out [M,N]
// MODE 1: f16 head-split  out[((b*16+h)*S + s)*64 + d] = (acc+bias)*alpha
// MODE 2: f16 head-split transposed (V): out[((b*16+h)*64 + d)*S + s]
template <int MODE>
__global__ __launch_bounds__(256) void gemm_nt(const _Float16* __restrict__ X,
                                               const _Float16* __restrict__ Wt,
                                               const float* __restrict__ bias,
                                               void* __restrict__ Out,
                                               float alpha) {
  __shared__ _Float16 sX[2][128 * 40];
  __shared__ _Float16 sW[2][128 * 40];

  const int tid  = threadIdx.x;
  const int lane = tid & 31;
  const int wave = tid >> 5;
  const int wm   = wave & 3;    // M sub-tile (32 rows)
  const int wn   = wave >> 2;   // N sub-tile (64 cols)
  const int r    = lane & 15;
  const int hi   = lane >> 4;
  const int m0   = blockIdx.y * 128;
  const int n0   = blockIdx.x * 128;

  const v8f zero = {0.f,0.f,0.f,0.f,0.f,0.f,0.f,0.f};
  v8f acc[2][4];
#pragma unroll
  for (int i = 0; i < 2; ++i)
#pragma unroll
    for (int j = 0; j < 4; ++j) acc[i][j] = zero;

  const int ldRow = tid >> 1;
  const int ldCol = (tid & 1) * 16;
  const _Float16* gXbase = X  + (size_t)(m0 + ldRow) * DM + ldCol;
  const _Float16* gWbase = Wt + (size_t)(n0 + ldRow) * DM + ldCol;
  const int ldsOff = ldRow * 40 + ldCol;

  // preload k0 = 0 into buffer 0
  {
#if HAS_ASYNC
    async_copy16(gXbase,     sX[0] + ldsOff);
    async_copy16(gXbase + 8, sX[0] + ldsOff + 8);
    async_copy16(gWbase,     sW[0] + ldsOff);
    async_copy16(gWbase + 8, sW[0] + ldsOff + 8);
    wait_async0();
#else
    *(uint4*)(sX[0] + ldsOff)     = *(const uint4*)(gXbase);
    *(uint4*)(sX[0] + ldsOff + 8) = *(const uint4*)(gXbase + 8);
    *(uint4*)(sW[0] + ldsOff)     = *(const uint4*)(gWbase);
    *(uint4*)(sW[0] + ldsOff + 8) = *(const uint4*)(gWbase + 8);
#endif
  }
  __syncthreads();

  int cur = 0;
  for (int k0 = 0; k0 < DM; k0 += 32, cur ^= 1) {
    const int kn = k0 + 32;
    if (kn < DM) {
      const _Float16* gX = gXbase + kn;
      const _Float16* gW = gWbase + kn;
      _Float16* lX = sX[cur ^ 1] + ldsOff;
      _Float16* lW = sW[cur ^ 1] + ldsOff;
#if HAS_ASYNC
      async_copy16(gX,     lX);
      async_copy16(gX + 8, lX + 8);
      async_copy16(gW,     lW);
      async_copy16(gW + 8, lW + 8);
#else
      *(uint4*)lX       = *(const uint4*)gX;
      *(uint4*)(lX + 8) = *(const uint4*)(gX + 8);
      *(uint4*)lW       = *(const uint4*)gW;
      *(uint4*)(lW + 8) = *(const uint4*)(gW + 8);
#endif
    }

    v16h a[2], b[4];
#pragma unroll
    for (int i = 0; i < 2; ++i) {
      const _Float16* p = sX[cur] + (wm * 32 + i * 16 + r) * 40;
      a[i] = ld16(p + 8 * hi, p + 16 + 8 * hi);
    }
#pragma unroll
    for (int j = 0; j < 4; ++j) {
      const _Float16* p = sW[cur] + (wn * 64 + j * 16 + r) * 40;
      b[j] = ld16(p + 8 * hi, p + 16 + 8 * hi);
    }
#pragma unroll
    for (int i = 0; i < 2; ++i)
#pragma unroll
      for (int j = 0; j < 4; ++j)
        acc[i][j] = __builtin_amdgcn_wmma_f32_16x16x32_f16(
            false, a[i], false, b[j], (short)0, acc[i][j], false, false);

#if HAS_ASYNC
    wait_async0();
#endif
    __syncthreads();
  }

#pragma unroll
  for (int i = 0; i < 2; ++i) {
#pragma unroll
    for (int j = 0; j < 4; ++j) {
      const int n  = n0 + wn * 64 + j * 16 + r;
      const float bn = bias[n];
#pragma unroll
      for (int v = 0; v < 8; ++v) {
        const int m = m0 + wm * 32 + i * 16 + hi * 8 + v;
        const float val = acc[i][j][v] + bn;
        if constexpr (MODE == 0) {
          ((float*)Out)[(size_t)m * DM + n] = val;
        } else {
          const int bb = m >> 11, s = m & (SEQ - 1);
          const int h  = n >> 6,  d = n & (HEAD_DIM - 1);
          _Float16* O16 = (_Float16*)Out;
          if constexpr (MODE == 1)
            O16[(((size_t)(bb * NHEAD + h)) * SEQ + s) * HEAD_DIM + d] = (_Float16)(val * alpha);
          else
            O16[(((size_t)(bb * NHEAD + h)) * HEAD_DIM + d) * SEQ + s] = (_Float16)val;
        }
      }
    }
  }
}

// ---------------------------------------------------------------- attention
// Transposed formulation: S^T = K·Q^T, O^T = V^T·P^T.
// C-layout column == query == lane&15 -> per-lane scalar softmax stats,
// only one __shfl_xor(,16) per reduction. Q pre-scaled by log2(e)/sqrt(64).
__global__ __launch_bounds__(256) void attn_fa(const _Float16* __restrict__ Qh, // [B*H,S,64]
                                               const _Float16* __restrict__ Kh, // [B*H,S,64]
                                               const _Float16* __restrict__ Vt, // [B*H,64,S]
                                               _Float16* __restrict__ Oh) {     // [B,S,H*64]
  __shared__ _Float16 sP[8 * 16 * 32];   // per-wave P^T tile: [query][32 keys]

  const int bh   = blockIdx.y;               // 0..31
  const int b    = bh >> 4, h = bh & 15;
  const int tid  = threadIdx.x;
  const int lane = tid & 31;
  const int wave = tid >> 5;
  const int r    = lane & 15;
  const int hi   = lane >> 4;
  const int q0   = blockIdx.x * 128 + wave * 16;

  const _Float16* Qp = Qh + ((size_t)bh * SEQ + q0) * HEAD_DIM;
  const _Float16* Kp = Kh + (size_t)bh * SEQ * HEAD_DIM;
  const _Float16* Vp = Vt + (size_t)bh * HEAD_DIM * SEQ;
  _Float16* myP = sP + wave * (16 * 32);

  // Q B-fragments: col = query q0+r; element i -> d = t*32 + (i%8)+(i/8)*16+8*hi
  v16h qf[2];
#pragma unroll
  for (int t = 0; t < 2; ++t) {
    const _Float16* p = Qp + r * HEAD_DIM + t * 32;
    qf[t] = ld16(p + 8 * hi, p + 16 + 8 * hi);
  }

  const v8f zero = {0.f,0.f,0.f,0.f,0.f,0.f,0.f,0.f};
  v8f o[4];
#pragma unroll
  for (int jd = 0; jd < 4; ++jd) o[jd] = zero;
  float mrun = -1e30f, lrun = 0.f;

  for (int kt = 0; kt < SEQ; kt += 32) {
    // S^T tiles: M=key(16 per tile j), N=query, K-dim=d (2 steps of 32)
    v8f c[2];
#pragma unroll
    for (int j = 0; j < 2; ++j) {
      v8f cc = zero;
#pragma unroll
      for (int t = 0; t < 2; ++t) {
        const _Float16* p = Kp + (size_t)(kt + j * 16 + r) * HEAD_DIM + t * 32;
        v16h kf = ld16(p + 8 * hi, p + 16 + 8 * hi);
        cc = __builtin_amdgcn_wmma_f32_16x16x32_f16(false, kf, false, qf[t],
                                                    (short)0, cc, false, false);
      }
      c[j] = cc;
    }

    // V A-fragments: row d = jd*16+r; element i -> key = kt + (i%8)+(i/8)*16+8*hi
    // (issued early so the loads overlap the softmax VALU work)
    v16h vf[4];
#pragma unroll
    for (int jd = 0; jd < 4; ++jd) {
      const _Float16* p = Vp + (size_t)(jd * 16 + r) * SEQ + kt;
      vf[jd] = ld16(p + 8 * hi, p + 16 + 8 * hi);
    }

    // per-lane online softmax for query q0+r (partner lane = lane^16)
    float lm = c[0][0];
#pragma unroll
    for (int v = 1; v < 8; ++v) lm = fmaxf(lm, c[0][v]);
#pragma unroll
    for (int v = 0; v < 8; ++v) lm = fmaxf(lm, c[1][v]);
    lm = fmaxf(lm, __shfl_xor(lm, 16, 32));
    const float mnew  = fmaxf(mrun, lm);
    const float scale = fast_exp2(mrun - mnew);
    float ls = 0.f;
    v8h ph[2];
#pragma unroll
    for (int j = 0; j < 2; ++j)
#pragma unroll
      for (int v = 0; v < 8; ++v) {
        const float pv = fast_exp2(c[j][v] - mnew);
        ls += pv;
        ph[j][v] = (_Float16)pv;
      }
    ls += __shfl_xor(ls, 16, 32);
    lrun = lrun * scale + ls;
    mrun = mnew;
#pragma unroll
    for (int jd = 0; jd < 4; ++jd)
#pragma unroll
      for (int v = 0; v < 8; ++v) o[jd][v] *= scale;

    // P^T -> LDS row q=r: tile j covers keys 16j + 8hi + [0..7]  (two 16B stores)
    *(v8h*)(myP + r * 32 + 8 * hi)      = ph[0];
    *(v8h*)(myP + r * 32 + 16 + 8 * hi) = ph[1];

    // P^T B-frag: col = query r; element i -> key = (i%8)+(i/8)*16+8*hi
    const _Float16* pp = myP + r * 32;
    v16h pf = ld16(pp + 8 * hi, pp + 16 + 8 * hi);

    // O^T += V^T(16d x 32k) @ P^T(32k x 16q)
#pragma unroll
    for (int jd = 0; jd < 4; ++jd)
      o[jd] = __builtin_amdgcn_wmma_f32_16x16x32_f16(false, vf[jd], false, pf,
                                                     (short)0, o[jd], false, false);
  }

  // O^T: row d = jd*16 + 8*hi + v, col query = r  -> vectorized 16B stores
  const float inv = 1.0f / lrun;
  const int s = q0 + r;
#pragma unroll
  for (int jd = 0; jd < 4; ++jd) {
    v8h ov;
#pragma unroll
    for (int v = 0; v < 8; ++v) ov[v] = (_Float16)(o[jd][v] * inv);
    *(v8h*)(Oh + ((size_t)b * SEQ + s) * DM + h * HEAD_DIM + jd * 16 + 8 * hi) = ov;
  }
}

// ---------------------------------------------------------------- launch
extern "C" void kernel_launch(void* const* d_in, const int* in_sizes, int n_in,
                              void* d_out, int out_size, void* d_ws, size_t ws_size,
                              hipStream_t stream) {
  (void)in_sizes; (void)n_in; (void)out_size; (void)ws_size;
  const float* x  = (const float*)d_in[0];
  const float* Wq = (const float*)d_in[1];
  const float* bq = (const float*)d_in[2];
  const float* Wk = (const float*)d_in[3];
  const float* bk = (const float*)d_in[4];
  const float* Wv = (const float*)d_in[5];
  const float* bv = (const float*)d_in[6];
  const float* Wo = (const float*)d_in[7];
  const float* bo = (const float*)d_in[8];

  char* ws = (char*)d_ws;
  const size_t MB = 1u << 20;
  _Float16* xh  = (_Float16*)(ws);             // 8 MB, reused as attn-out later
  _Float16* Wqt = (_Float16*)(ws +  8 * MB);   // 2 MB each
  _Float16* Wkt = (_Float16*)(ws + 10 * MB);
  _Float16* Wvt = (_Float16*)(ws + 12 * MB);
  _Float16* Wot = (_Float16*)(ws + 14 * MB);
  _Float16* Qh  = (_Float16*)(ws + 16 * MB);   // 8 MB each
  _Float16* Kh  = (_Float16*)(ws + 24 * MB);
  _Float16* Vt  = (_Float16*)(ws + 32 * MB);
  _Float16* Ah  = xh;                          // reuse x-f16 buffer for attn out

  cvt_f32_to_f16<<<2048, 256, 0, stream>>>(x, xh, MROWS * DM);

  dim3 tb(32, 8), tg(DM / 32, DM / 32);
  transpose_w<<<tg, tb, 0, stream>>>(Wq, Wqt);
  transpose_w<<<tg, tb, 0, stream>>>(Wk, Wkt);
  transpose_w<<<tg, tb, 0, stream>>>(Wv, Wvt);
  transpose_w<<<tg, tb, 0, stream>>>(Wo, Wot);

  const float alphaQ = 0.125f * 1.4426950408889634f;  // 1/sqrt(64) * log2(e)
  dim3 gg(DM / 128, MROWS / 128);
  gemm_nt<1><<<gg, 256, 0, stream>>>(xh, Wqt, bq, Qh, alphaQ);
  gemm_nt<1><<<gg, 256, 0, stream>>>(xh, Wkt, bk, Kh, 1.0f);
  gemm_nt<2><<<gg, 256, 0, stream>>>(xh, Wvt, bv, Vt, 1.0f);

  attn_fa<<<dim3(SEQ / 128, BATCH * NHEAD), 256, 0, stream>>>(Qh, Kh, Vt, Ah);

  gemm_nt<0><<<gg, 256, 0, stream>>>(Ah, Wot, bo, d_out, 1.0f);
}
